// Additive_Attention_44530220925449
// MI455X (gfx1250) — matrix operand order repaired and checked by reading, not compiled, not run on hardware
//
#include <hip/hip_runtime.h>
#include <hip/hip_bf16.h>

#define NEG_INF_F (-1000000000.0f)

typedef __attribute__((ext_vector_type(16))) __bf16 v16bf;
typedef __attribute__((ext_vector_type(8)))  float  v8f;
typedef __attribute__((ext_vector_type(4)))  float  v4f;

#if __has_builtin(__builtin_amdgcn_tanhf)
#define FAST_TANH(x) __builtin_amdgcn_tanhf(x)
#elif __has_builtin(__builtin_amdgcn_tanh_f32)
#define FAST_TANH(x) __builtin_amdgcn_tanh_f32(x)
#else
#define FAST_TANH(x) tanhf(x)
#endif

// Problem constants (reference: B=4, H=1, Lq=Lkv=256, D=512)
constexpr int B_   = 4;
constexpr int LQ   = 256;
constexpr int LKV  = 256;
constexpr int D_   = 512;
constexpr int NROW = B_ * LQ;          // 1024 flattened (b,h,q) rows
constexpr int QT   = 4;                // q rows per block (k_proj reuse factor)

// ---------------------------------------------------------------------------
// Kernel 1: projections via bf16 WMMA (unchanged from round 1).
//   P[0][n][e] = sum_d Q[n][d] * Wc[e][d]        + bc[e]   (q_proj, bc folded)
//   P[1][n][e] = sum_d K[n][d] * Wc[e][512 + d]            (k_proj)
// ---------------------------------------------------------------------------
__global__ __launch_bounds__(128)
void aa_proj_kernel(const float* __restrict__ Q,
                    const float* __restrict__ Km,
                    const float* __restrict__ Wc,
                    const float* __restrict__ bc,
                    float* __restrict__ P)
{
    const int z    = blockIdx.z;                 // 0 = q-proj, 1 = k-proj
    const int wave = threadIdx.x >> 5;
    const int lane = threadIdx.x & 31;
    const int tile = blockIdx.x * 4 + wave;      // 0..2047
    const int tm   = tile >> 5;                  // row tile 0..63
    const int tn   = tile & 31;                  // col tile 0..31

    const float* __restrict__ X = (z == 0) ? Q : Km;
    const int wofs = z * D_;

    const int  hlane  = lane & 15;
    const bool hiHalf = (lane >= 16);

    const float* pa_row = X + (size_t)(tm * 16 + hlane) * D_;
    const int a_koff = hiHalf ? 8 : 0;           // A: lanes 0-15 K{0-7,16-23}; 16-31 K{8-15,24-31}

    const float* pb_row = Wc + (size_t)(tn * 16 + hlane) * (2 * D_) + wofs;
    const int b_koff = hiHalf ? 8 : 0;           // B: lanes 0-15 K{0-7,16-23}; 16-31 K{8-15,24-31}

    v8f c = {};
    for (int kb = 0; kb < D_; kb += 32) {
        v16bf a, b;
        const float* pa = pa_row + kb + a_koff;
        #pragma unroll
        for (int i = 0; i < 8; ++i) a[i]     = (__bf16)pa[i];
        #pragma unroll
        for (int i = 0; i < 8; ++i) a[8 + i] = (__bf16)pa[16 + i];

        const float* pb = pb_row + kb + b_koff;
        #pragma unroll
        for (int i = 0; i < 8; ++i) b[i]     = (__bf16)pb[i];
        #pragma unroll
        for (int i = 0; i < 8; ++i) b[8 + i] = (__bf16)pb[16 + i];

        c = __builtin_amdgcn_wmma_f32_16x16x32_bf16(false, a, false, b,
                                                    (short)0, c, false, false);
    }

    const int n_out  = tn * 16 + hlane;
    const int m_base = tm * 16 + (hiHalf ? 8 : 0);
    const float badd = (z == 0) ? bc[n_out] : 0.0f;
    float* __restrict__ Pz = P + (size_t)z * NROW * D_;
    #pragma unroll
    for (int r = 0; r < 8; ++r)
        Pz[(size_t)(m_base + r) * D_ + n_out] = c[r] + badd;
}

// ---------------------------------------------------------------------------
// Kernel 2: logits + log-softmax. One block = QT q rows of one b.
// Lane owns the 16 d-indices { 128*c + 4*lane + e : c<4, e<4 }.
// q_proj (QT rows) and Wl live entirely in registers; k_proj is read once
// per block via global_load_b128 and reused across QT q rows.
// ---------------------------------------------------------------------------
__global__ __launch_bounds__(256)
void aa_logits_kernel(const float* __restrict__ P,
                      const int*   __restrict__ mask,
                      const float* __restrict__ Wl,
                      const float* __restrict__ bl,
                      float* __restrict__ out)
{
    __shared__ float logits[QT][LKV];
    __shared__ float red[8];

    const int tile = blockIdx.x;            // 0..255
    const int b    = tile / (LQ / QT);      // 64 tiles per batch
    const int q0   = (tile % (LQ / QT)) * QT;
    const int t    = threadIdx.x;           // 0..255
    const int wave = t >> 5;
    const int lane = t & 31;

    // register-resident q_proj rows (bc folded in) and Wl
    v4f qv[QT][4];
    v4f wv[4];
    const float* __restrict__ qbase = P + (size_t)(b * LQ + q0) * D_;
    #pragma unroll
    for (int qi = 0; qi < QT; ++qi)
        #pragma unroll
        for (int c = 0; c < 4; ++c)
            qv[qi][c] = *(const v4f*)(qbase + (size_t)qi * D_ + c * 128 + 4 * lane);
    #pragma unroll
    for (int c = 0; c < 4; ++c)
        wv[c] = *(const v4f*)(Wl + c * 128 + 4 * lane);

    const float blv = bl[0];
    const float* __restrict__ Pk = P + (size_t)NROW * D_ + (size_t)b * LKV * D_;

    // hot loop: each wave owns k = wave + 8*j; 4x b128 loads feed QT*16 tanh+fma
    for (int j = 0; j < LKV / 8; ++j) {
        const int k = j * 8 + wave;
        const float* __restrict__ kp = Pk + (size_t)k * D_;

        v4f kv[4];
        #pragma unroll
        for (int c = 0; c < 4; ++c)
            kv[c] = *(const v4f*)(kp + c * 128 + 4 * lane);

        float acc[QT] = {0.0f, 0.0f, 0.0f, 0.0f};
        #pragma unroll
        for (int c = 0; c < 4; ++c)
            #pragma unroll
            for (int e = 0; e < 4; ++e) {
                const float kd = kv[c][e];
                const float wd = wv[c][e];
                #pragma unroll
                for (int qi = 0; qi < QT; ++qi)
                    acc[qi] += FAST_TANH(qv[qi][c][e] + kd) * wd;
            }

        #pragma unroll
        for (int qi = 0; qi < QT; ++qi)
            #pragma unroll
            for (int o = 16; o > 0; o >>= 1)
                acc[qi] += __shfl_xor(acc[qi], o, 32);

        if (lane == 0) {
            const float mb = (mask[b * LKV + k] == 0) ? NEG_INF_F : 1.0f;
            #pragma unroll
            for (int qi = 0; qi < QT; ++qi)
                logits[qi][k] = acc[qi] + blv + mb;
        }
    }
    __syncthreads();

    // log-softmax per q row (thread t owns column t)
    for (int qi = 0; qi < QT; ++qi) {
        const float x = logits[qi][t];

        float m = x;
        #pragma unroll
        for (int o = 16; o > 0; o >>= 1) m = fmaxf(m, __shfl_xor(m, o, 32));
        if (lane == 0) red[wave] = m;
        __syncthreads();
        m = red[0];
        #pragma unroll
        for (int w = 1; w < 8; ++w) m = fmaxf(m, red[w]);
        __syncthreads();                   // before reusing red[]

        float s = __expf(x - m);
        #pragma unroll
        for (int o = 16; o > 0; o >>= 1) s += __shfl_xor(s, o, 32);
        if (lane == 0) red[wave] = s;
        __syncthreads();
        s = 0.0f;
        #pragma unroll
        for (int w = 0; w < 8; ++w) s += red[w];
        __syncthreads();                   // before next qi overwrites red[]

        out[(size_t)(b * LQ + q0 + qi) * LKV + t] = (x - m) - __logf(s);
    }
}

// ---------------------------------------------------------------------------
extern "C" void kernel_launch(void* const* d_in, const int* in_sizes, int n_in,
                              void* d_out, int out_size, void* d_ws, size_t ws_size,
                              hipStream_t stream) {
    const float* queries = (const float*)d_in[0];   // (4,1,256,512) f32
    const float* keys    = (const float*)d_in[1];   // (4,1,256,512) f32
    // d_in[2] = values: unused by the reference
    const int*   mask    = (const int*)d_in[3];     // (4,256) i32
    const float* Wc      = (const float*)d_in[4];   // (512,1024) f32
    const float* bc      = (const float*)d_in[5];   // (512,) f32
    const float* Wl      = (const float*)d_in[6];   // (1,512) f32
    const float* bl      = (const float*)d_in[7];   // (1,) f32
    float* out = (float*)d_out;                     // (4,256,256) f32
    float* P   = (float*)d_ws;                      // 2*1024*512 f32 = 4 MB scratch

    dim3 pgrid(512, 1, 2);                          // 2048 tiles per matrix, 4 waves/block
    aa_proj_kernel<<<pgrid, 128, 0, stream>>>(queries, keys, Wc, bc, P);
    aa_logits_kernel<<<dim3(B_ * (LQ / QT)), 256, 0, stream>>>(P, mask, Wl, bl, out);
}
